// UpsampleWithRefrence_23502061044009
// MI455X (gfx1250) — matrix operand order repaired and checked by reading, not compiled, not run on hardware
//
#include <hip/hip_runtime.h>
#include <math.h>

typedef __attribute__((ext_vector_type(16))) __bf16 v16bf;
typedef __attribute__((ext_vector_type(8)))  float  v8f;

#define H_LO 128
#define W_LO 128
#define H_HI 256
#define W_HI 256
#define CLOW 64
#define CMID 128
#define CHI  16
#define NIMG 16
#define NREF 32          // 2*CHI outputs of conv2
#define XPITCH 66        // bf16 halves per pixel row in LDS (33 dwords, conflict-free)
#define WPITCH 66
#define SPITCH 34        // 17 dwords, conflict-free

__device__ __forceinline__ float gelu_exact(float v) {
    return 0.5f * v * (1.0f + erff(v * 0.70710678118654752440f));
}

// ---------------------------------------------------------------------------
// Upsample one 128-pixel half-row tile of x = bilinear(lowres) into LDS (bf16)
// and load conv1 weights (bf16). 256 threads.
// ---------------------------------------------------------------------------
__device__ __forceinline__ void build_tiles(const float* __restrict__ lowres,
                                            const float* __restrict__ w1,
                                            __bf16* xs, __bf16* ws,
                                            int img, int i, int j0, int t) {
    // conv1 weights -> LDS bf16, padded rows
    for (int idx = t; idx < CMID * CLOW; idx += 256) {
        int row = idx >> 6, k = idx & 63;
        ws[row * WPITCH + k] = (__bf16)w1[row * CLOW + k];
    }
    // bilinear upsample, align_corners=True: scale = 127/255
    const float sc = 127.0f / 255.0f;
    float fy = (float)i * sc;
    int   y0 = (int)floorf(fy);
    float wy = fy - (float)y0;
    int   y1 = min(y0 + 1, H_LO - 1);

    int p  = t & 127;           // pixel within tile
    int c0 = t >> 7;            // 0 or 1
    int j  = j0 + p;
    float fx = (float)j * sc;
    int   x0 = (int)floorf(fx);
    float wx = fx - (float)x0;
    int   x1 = min(x0 + 1, W_LO - 1);
    float w00 = (1.f - wy) * (1.f - wx), w01 = (1.f - wy) * wx;
    float w10 = wy * (1.f - wx),          w11 = wy * wx;
    const float* base = lowres + (size_t)img * CLOW * H_LO * W_LO;
    for (int c = c0; c < CLOW; c += 2) {
        const float* pc = base + (size_t)c * H_LO * W_LO;
        float v = w00 * pc[y0 * W_LO + x0] + w01 * pc[y0 * W_LO + x1]
                + w10 * pc[y1 * W_LO + x0] + w11 * pc[y1 * W_LO + x1];
        xs[p * XPITCH + c] = (__bf16)v;
    }
}

// ---------------------------------------------------------------------------
// Kernel 1: stats pass.  r1 = gelu(conv1(upsample(lowres))); accumulate
// per-channel sums and per-group sum-of-squares (for GN + SE fold).
// grid = (2, 256, 16), block = 256 (8 waves, 16 pixels each).
// ---------------------------------------------------------------------------
__global__ __launch_bounds__(256) void k_stats(const float* __restrict__ lowres,
                                               const float* __restrict__ w1,
                                               const float* __restrict__ b1,
                                               float* __restrict__ gsum,
                                               float* __restrict__ gsq) {
    __shared__ __bf16 xs[128 * XPITCH];
    __shared__ __bf16 ws[CMID * WPITCH];
    __shared__ float  lsum[CMID];
    __shared__ float  lsq[2];

    const int t   = threadIdx.x;
    const int img = blockIdx.z;
    const int i   = blockIdx.y;
    const int j0  = blockIdx.x * 128;

    if (t < CMID) lsum[t] = 0.f;
    if (t < 2)    lsq[t]  = 0.f;
    build_tiles(lowres, w1, xs, ws, img, i, j0, t);
    __syncthreads();

    const int lane    = t & 31;
    const int m16     = lane & 15;
    const int hi16    = lane >> 4;
    const int pixBase = (t >> 5) * 16;

    // A fragments (16 px x 32 ch), kc = 0,1
    v16bf afrag[2];
    {
        const __bf16* xp = &xs[(pixBase + m16) * XPITCH];
        #pragma unroll
        for (int kc = 0; kc < 2; ++kc) {
            int kb = kc * 32 + (hi16 ? 16 : 0);
            #pragma unroll
            for (int h = 0; h < 8; ++h) {
                afrag[kc][h]     = xp[kb + h];
                afrag[kc][8 + h] = xp[kb + 8 + h];
            }
        }
    }

    for (int nc = 0; nc < 4; ++nc) {          // 4 chunks of 32 output channels
        v8f acc[2] = {v8f{}, v8f{}};
        #pragma unroll
        for (int kc = 0; kc < 2; ++kc) {
            int kb = kc * 32 + (hi16 ? 16 : 0);
            #pragma unroll
            for (int nt = 0; nt < 2; ++nt) {
                v16bf b;
                const __bf16* wp = &ws[(nc * 32 + nt * 16 + m16) * WPITCH + kb];
                #pragma unroll
                for (int h = 0; h < 16; ++h) b[h] = wp[h];
                acc[nt] = __builtin_amdgcn_wmma_f32_16x16x32_bf16(
                    false, afrag[kc], false, b, (short)0, acc[nt], false, false);
            }
        }
        #pragma unroll
        for (int nt = 0; nt < 2; ++nt) {
            int   c  = nc * 32 + nt * 16 + m16;
            float bb = b1[c];
            float s8 = 0.f, q8 = 0.f;
            #pragma unroll
            for (int r = 0; r < 8; ++r) {
                float v = gelu_exact(acc[nt][r] + bb);
                s8 += v; q8 += v * v;
            }
            atomicAdd(&lsum[c], s8);
            atomicAdd(&lsq[c >> 6], q8);
        }
    }
    __syncthreads();
    if (t < CMID) atomicAdd(&gsum[img * CMID + t], lsum[t]);
    if (t < 2)    atomicAdd(&gsq[img * 2 + t], lsq[t]);
}

// ---------------------------------------------------------------------------
// Kernel 2: fold GN + SE into conv2:  W2'[o,c] = conv2_w[o,c]*A_c (bf16),
// bias2[o] = sum_c conv2_w[o,c]*B_c.  One block, 128 threads.
// ---------------------------------------------------------------------------
__global__ __launch_bounds__(128) void k_fold(const float* __restrict__ gsum,
                                              const float* __restrict__ gsq,
                                              const float* __restrict__ gn_w,
                                              const float* __restrict__ gn_b,
                                              const float* __restrict__ se_w1,
                                              const float* __restrict__ se_w2,
                                              const float* __restrict__ conv2_w,
                                              unsigned short* __restrict__ w2p_raw,
                                              float* __restrict__ bias2) {
    __bf16* w2p = reinterpret_cast<__bf16*>(w2p_raw);
    __shared__ float m[CMID], Af[CMID], Bf[CMID], y1s[16], mus[2], rss[2];
    const int t = threadIdx.x;
    const float Npix = 65536.0f;

    for (int img = 0; img < NIMG; ++img) {
        float S = gsum[img * CMID + t];
        m[t] = S;
        __syncthreads();
        if (t < 2) {
            float a = 0.f;
            for (int c = 0; c < 64; ++c) a += m[t * 64 + c];
            float Ng  = 64.0f * Npix;
            float mu  = a / Ng;
            float var = gsq[img * 2 + t] / Ng - mu * mu;
            mus[t] = mu;
            rss[t] = rsqrtf(var + 1e-5f);
        }
        __syncthreads();
        int   g   = t >> 6;
        float mm  = (S / Npix - mus[g]) * rss[g] * gn_w[t] + gn_b[t]; // post-GN chan mean
        m[t] = mm;
        __syncthreads();
        if (t < 16) {
            float a = 0.f;
            for (int c = 0; c < CMID; ++c) a += se_w1[t * CMID + c] * m[c];
            y1s[t] = fmaxf(a, 0.f);
        }
        __syncthreads();
        float y2 = 0.f;
        for (int j = 0; j < 16; ++j) y2 += se_w2[t * 16 + j] * y1s[j];
        float sg = 1.0f / (1.0f + expf(-y2));
        Af[t] = rss[g] * gn_w[t] * sg;
        Bf[t] = (gn_b[t] - mus[g] * rss[g] * gn_w[t]) * sg;
        __syncthreads();
        for (int o = 0; o < NREF; ++o)
            w2p[((size_t)img * NREF + o) * CMID + t] =
                (__bf16)(conv2_w[o * CMID + t] * Af[t]);
        if (t < NREF) {
            float a = 0.f;
            for (int c = 0; c < CMID; ++c) a += conv2_w[t * CMID + c] * Bf[c];
            bias2[img * NREF + t] = a;
        }
        __syncthreads();
    }
}

// ---------------------------------------------------------------------------
// Kernel 3: main pass.  Recompute r1 = gelu(conv1(x)) per 32-channel chunk,
// stage bf16 in LDS, apply folded conv2 via second WMMA chain, write ref.
// ---------------------------------------------------------------------------
__global__ __launch_bounds__(256) void k_main(const float* __restrict__ lowres,
                                              const float* __restrict__ w1,
                                              const float* __restrict__ b1,
                                              const unsigned short* __restrict__ w2p_raw,
                                              const float* __restrict__ bias2,
                                              float* __restrict__ refbuf) {
    const __bf16* w2p = reinterpret_cast<const __bf16*>(w2p_raw);
    __shared__ __bf16 xs[128 * XPITCH];
    __shared__ __bf16 ws[CMID * WPITCH];
    __shared__ __bf16 stage[8 * 16 * SPITCH];

    const int t   = threadIdx.x;
    const int img = blockIdx.z;
    const int i   = blockIdx.y;
    const int j0  = blockIdx.x * 128;

    build_tiles(lowres, w1, xs, ws, img, i, j0, t);
    __syncthreads();

    const int lane    = t & 31;
    const int m16     = lane & 15;
    const int hi16    = lane >> 4;
    const int wv      = t >> 5;
    const int pixBase = wv * 16;
    __bf16* st = &stage[wv * 16 * SPITCH];

    v16bf afrag[2];
    {
        const __bf16* xp = &xs[(pixBase + m16) * XPITCH];
        #pragma unroll
        for (int kc = 0; kc < 2; ++kc) {
            int kb = kc * 32 + (hi16 ? 16 : 0);
            #pragma unroll
            for (int h = 0; h < 8; ++h) {
                afrag[kc][h]     = xp[kb + h];
                afrag[kc][8 + h] = xp[kb + 8 + h];
            }
        }
    }

    // ref accumulators (32 outputs = 2 N-tiles), init with folded bias
    v8f racc[2];
    {
        float bo0 = bias2[img * NREF + m16];
        float bo1 = bias2[img * NREF + 16 + m16];
        #pragma unroll
        for (int r = 0; r < 8; ++r) { racc[0][r] = bo0; racc[1][r] = bo1; }
    }

    for (int nc = 0; nc < 4; ++nc) {          // 32 mid-channels per chunk
        // ---- conv1 for this chunk ----
        v8f acc[2] = {v8f{}, v8f{}};
        #pragma unroll
        for (int kc = 0; kc < 2; ++kc) {
            int kb = kc * 32 + (hi16 ? 16 : 0);
            #pragma unroll
            for (int nt = 0; nt < 2; ++nt) {
                v16bf b;
                const __bf16* wp = &ws[(nc * 32 + nt * 16 + m16) * WPITCH + kb];
                #pragma unroll
                for (int h = 0; h < 16; ++h) b[h] = wp[h];
                acc[nt] = __builtin_amdgcn_wmma_f32_16x16x32_bf16(
                    false, afrag[kc], false, b, (short)0, acc[nt], false, false);
            }
        }
        // ---- bias + GELU, stage to LDS (per-wave region) ----
        #pragma unroll
        for (int nt = 0; nt < 2; ++nt) {
            int   cl = nt * 16 + m16;          // channel within chunk
            float bb = b1[nc * 32 + cl];
            #pragma unroll
            for (int r = 0; r < 8; ++r) {
                float v = gelu_exact(acc[nt][r] + bb);
                st[(r + 8 * hi16) * SPITCH + cl] = (__bf16)v;
            }
        }
        // ---- second GEMM: K-chunk of 32, folded conv2 ----
        v16bf a2;
        {
            const __bf16* sp = &st[m16 * SPITCH];
            int kb = hi16 ? 16 : 0;
            #pragma unroll
            for (int h = 0; h < 8; ++h) {
                a2[h]     = sp[kb + h];
                a2[8 + h] = sp[kb + 8 + h];
            }
        }
        #pragma unroll
        for (int nt = 0; nt < 2; ++nt) {
            v16bf b2;
            const __bf16* wp =
                &w2p[((size_t)img * NREF + nt * 16 + m16) * CMID + nc * 32 + hi16 * 16];
            #pragma unroll
            for (int h = 0; h < 16; ++h) b2[h] = wp[h];
            racc[nt] = __builtin_amdgcn_wmma_f32_16x16x32_bf16(
                false, a2, false, b2, (short)0, racc[nt], false, false);
        }
    }

    // write ref: lane holds 8 consecutive pixels for one output channel
    const int jb = j0 + pixBase + 8 * hi16;
    #pragma unroll
    for (int nt = 0; nt < 2; ++nt) {
        int o = nt * 16 + m16;
        float* op = refbuf + (((size_t)img * NREF + o) * H_HI + i) * W_HI + jb;
        float4 lo = make_float4(racc[nt][0], racc[nt][1], racc[nt][2], racc[nt][3]);
        float4 hi = make_float4(racc[nt][4], racc[nt][5], racc[nt][6], racc[nt][7]);
        *reinterpret_cast<float4*>(op)     = lo;
        *reinterpret_cast<float4*>(op + 4) = hi;
    }
}

// ---------------------------------------------------------------------------
// Kernel 4a: column-wise softmax+cumsum over plane (2k+1); write x-coordinate
// (in pixels) in place.  One thread per column; ref stays L2-resident.
// ---------------------------------------------------------------------------
__global__ __launch_bounds__(256) void k_col(float* __restrict__ refbuf) {
    const int j   = threadIdx.x;
    const int k   = blockIdx.x;
    const int img = blockIdx.y;
    float* p = refbuf + ((size_t)img * NREF + 2 * k + 1) * (size_t)(H_HI * W_HI);
    float mx = -3.0e38f;
    for (int i = 0; i < H_HI; ++i) mx = fmaxf(mx, p[i * W_HI + j]);
    float s = 0.f;
    for (int i = 0; i < H_HI; ++i) s += expf(p[i * W_HI + j] - mx);
    float inv = 1.0f / s, run = 0.f;
    for (int i = 0; i < H_HI; ++i) {
        run += expf(p[i * W_HI + j] - mx) * inv;
        p[i * W_HI + j] = run * (float)(W_HI - 1);   // sample x in pixels
    }
}

// ---------------------------------------------------------------------------
// Kernel 4b: per-row softmax+cumsum over plane (2k) -> sample y; read sample x
// from plane (2k+1); fused bilinear grid-sample of highres (zeros padding).
// ---------------------------------------------------------------------------
__global__ __launch_bounds__(256) void k_row_sample(const float* __restrict__ refbuf,
                                                    const float* __restrict__ hires,
                                                    float* __restrict__ out) {
    __shared__ float sm[256];
    const int j   = threadIdx.x;
    const int i   = blockIdx.x;
    const int k   = blockIdx.y;
    const int img = blockIdx.z;

    const float* p0 = refbuf + (((size_t)img * NREF + 2 * k) * H_HI + i) * W_HI;
    const float* p1 = refbuf + (((size_t)img * NREF + 2 * k + 1) * H_HI + i) * W_HI;
    float v = p0[j];
    sm[j] = v; __syncthreads();
    for (int s = 128; s > 0; s >>= 1) {
        if (j < s) sm[j] = fmaxf(sm[j], sm[j + s]);
        __syncthreads();
    }
    float mx = sm[0]; __syncthreads();
    float e = expf(v - mx);
    sm[j] = e; __syncthreads();
    for (int off = 1; off < 256; off <<= 1) {       // inclusive Hillis-Steele scan
        float tv = (j >= off) ? sm[j - off] : 0.f;
        __syncthreads();
        sm[j] += tv;
        __syncthreads();
    }
    float gy = (sm[j] / sm[255]) * (float)(H_HI - 1);   // sample y in pixels
    float gx = p1[j];                                   // sample x in pixels

    float x0f = floorf(gx), y0f = floorf(gy);
    float wx = gx - x0f, wy = gy - y0f;
    int x0 = (int)x0f, y0 = (int)y0f;
    const float* src = hires + ((size_t)img * CHI + k) * (size_t)(H_HI * W_HI);
    auto g = [&](int xx, int yy) -> float {
        return (xx >= 0 && xx < W_HI && yy >= 0 && yy < H_HI)
                   ? src[yy * W_HI + xx] : 0.0f;
    };
    float res = (1.f - wx) * (1.f - wy) * g(x0, y0)
              + wx * (1.f - wy)         * g(x0 + 1, y0)
              + (1.f - wx) * wy         * g(x0, y0 + 1)
              + wx * wy                 * g(x0 + 1, y0 + 1);
    out[(((size_t)img * CHI + k) * H_HI + i) * W_HI + j] = res;
}

// ---------------------------------------------------------------------------
extern "C" void kernel_launch(void* const* d_in, const int* in_sizes, int n_in,
                              void* d_out, int out_size, void* d_ws, size_t ws_size,
                              hipStream_t stream) {
    const float* lowres  = (const float*)d_in[0];
    const float* highres = (const float*)d_in[1];
    const float* conv1_w = (const float*)d_in[2];
    const float* conv1_b = (const float*)d_in[3];
    const float* gn_w    = (const float*)d_in[4];
    const float* gn_b    = (const float*)d_in[5];
    const float* se_w1   = (const float*)d_in[6];
    const float* se_w2   = (const float*)d_in[7];
    const float* conv2_w = (const float*)d_in[8];
    float* out = (float*)d_out;

    char* ws = (char*)d_ws;
    float* refbuf = (float*)ws;                                   // 128 MB, L2-resident
    size_t off = (size_t)NIMG * NREF * H_HI * W_HI * sizeof(float);
    float* gsum  = (float*)(ws + off); off += (size_t)NIMG * CMID * sizeof(float);
    float* gsq   = (float*)(ws + off); off += (size_t)NIMG * 2 * sizeof(float);
    float* bias2 = (float*)(ws + off); off += (size_t)NIMG * NREF * sizeof(float);
    unsigned short* w2p = (unsigned short*)(ws + off);            // bf16 folded conv2

    hipMemsetAsync(gsum, 0, (size_t)(NIMG * CMID + NIMG * 2) * sizeof(float), stream);

    k_stats<<<dim3(2, H_HI, NIMG), 256, 0, stream>>>(lowres, conv1_w, conv1_b, gsum, gsq);
    k_fold<<<1, 128, 0, stream>>>(gsum, gsq, gn_w, gn_b, se_w1, se_w2, conv2_w, w2p, bias2);
    k_main<<<dim3(2, H_HI, NIMG), 256, 0, stream>>>(lowres, conv1_w, conv1_b, w2p, bias2, refbuf);
    k_col<<<dim3(CHI, NIMG), 256, 0, stream>>>(refbuf);
    k_row_sample<<<dim3(H_HI, CHI, NIMG), 256, 0, stream>>>(refbuf, highres, out);
}
